// QuantizedLinear_59949153517749
// MI455X (gfx1250) — compile-verified, matrix operand order checked by reading, not run on hardware
//
#include <hip/hip_runtime.h>

// ---------------------------------------------------------------------------
// Quantized linear  Y = scale * (X @ Wq^T) + bias   on gfx1250 (CDNA5, wave32)
//   X  : [M=8192, K=4096]  fp32
//   Wq : [N=11008, K=4096] int8 values stored as int32
//   Y  : [M, N] fp32
// int8 weights are EXACT in bf16 (|q| <= 128 fits the 8-bit significand), so
// the weight pack uses plain truncation (exact for zp=0). Activations use RNE.
// Accumulate f32 with v_wmma_f32_16x16x32_bf16; scale+bias fold in epilogue.
// Register-double-buffered pipeline: global loads of tile kt+1 overlap the
// WMMAs of tile kt; LDS staging tile is single-buffered with split barriers.
// ---------------------------------------------------------------------------

typedef __attribute__((ext_vector_type(16))) __bf16 v16bf;
typedef __attribute__((ext_vector_type(8)))  float  v8f;

#define BM 128
#define BN 128
#define BK 32
#define LDS_STRIDE 20   // uints per row: 16 data (32 bf16) + 4 pad (bank spread)

// Round-to-nearest-even bf16, result sits in the HIGH 16 bits of the u32.
__device__ __forceinline__ unsigned int bf16rne_hi(float f) {
    unsigned int u = __builtin_bit_cast(unsigned int, f);
    return u + 0x7FFFu + ((u >> 16) & 1u);          // v_bfe + v_add3
}
// RNE pack of two floats into {hi.bf16, lo.bf16}: one v_perm_b32.
__device__ __forceinline__ unsigned int pack2_rne(float lo, float hi) {
    return __builtin_amdgcn_perm(bf16rne_hi(hi), bf16rne_hi(lo), 0x07060302u);
}
// Truncation pack (exact when inputs are already bf16-representable, e.g.
// int8-valued weights with zero_point = 0): one v_perm_b32, no rounding ops.
__device__ __forceinline__ unsigned int pack2_trunc(float lo, float hi) {
    return __builtin_amdgcn_perm(__builtin_bit_cast(unsigned int, hi),
                                 __builtin_bit_cast(unsigned int, lo),
                                 0x07060302u);
}

union Frag { v16bf v; uint4 q[2]; };

__global__ __launch_bounds__(256)
void qlinear_wmma_bf16(const float* __restrict__ X,
                       const int*   __restrict__ Wq,
                       const float* __restrict__ scale_p,
                       const float* __restrict__ zp_p,
                       const float* __restrict__ bias,
                       float*       __restrict__ Y)
{
    constexpr int K = 4096;
    constexpr int N = 11008;
    constexpr int kTiles = K / BK;                  // 128

    __shared__ unsigned int lA[BM * LDS_STRIDE];    // 128 x 32 bf16 (+pad) = 10 KB
    __shared__ unsigned int lB[BN * LDS_STRIDE];    // 128 x 32 bf16 (+pad) = 10 KB

    const int tid  = threadIdx.x;
    const int wave = tid >> 5;
    const int lane = tid & 31;
    const int half = lane >> 4;                     // K-half selector in a fragment
    const int l16  = lane & 15;                     // row within a 16-wide fragment

    const int m0 = blockIdx.y * BM;
    const int n0 = blockIdx.x * BN;

    const float scale = scale_p[0];
    const float zp    = zp_p[0];

    // 8 waves in a 2(M) x 4(N) grid; each wave owns a 64x32 output tile
    const int wm = (wave >> 2) * 64;
    const int wn = (wave & 3) * 32;

    const v8f vzero = {0.f, 0.f, 0.f, 0.f, 0.f, 0.f, 0.f, 0.f};
    v8f acc[4][2];
    #pragma unroll
    for (int i = 0; i < 4; ++i)
        #pragma unroll
        for (int j = 0; j < 2; ++j)
            acc[i][j] = vzero;

    // staging coordinates: each thread moves 4 rows x 4 K-elements per operand
    const int kq  = tid & 7;                        // K sub-offset kq*4
    const int row = tid >> 3;                       // base row 0..31, stride 32

    const float* xp = X  + (size_t)(m0 + row) * K + kq * 4;
    const int*   wp = Wq + (size_t)(n0 + row) * K + kq * 4;

    // ---- preload tile 0 into registers (8x global_load_b128, clause) ----
    float4 ax[4];
    int4   wx[4];
    #pragma unroll
    for (int i = 0; i < 4; ++i) ax[i] = *(const float4*)(xp + i * 32 * K);
    #pragma unroll
    for (int i = 0; i < 4; ++i) wx[i] = *(const int4*)(wp + i * 32 * K);

    for (int kt = 0; kt < kTiles; ++kt) {
        // ---- convert staged registers -> bf16 LDS tile ----
        #pragma unroll
        for (int i = 0; i < 4; ++i) {
            const int off = (row + 32 * i) * LDS_STRIDE + kq * 2;
            uint2 va;
            va.x = pack2_rne(ax[i].x, ax[i].y);
            va.y = pack2_rne(ax[i].z, ax[i].w);
            *(uint2*)&lA[off] = va;                 // ds_store_b64
            uint2 vb;                               // exact: int8 in bf16, zp = 0
            vb.x = pack2_trunc((float)wx[i].x - zp, (float)wx[i].y - zp);
            vb.y = pack2_trunc((float)wx[i].z - zp, (float)wx[i].w - zp);
            *(uint2*)&lB[off] = vb;                 // ds_store_b64
        }
        __syncthreads();

        // ---- fire global loads for tile kt+1 NOW; they complete under the
        //      WMMAs below (waits land at the top of the next iteration) ----
        if (kt + 1 < kTiles) {
            const float* xk = xp + (kt + 1) * BK;
            const int*   wk = wp + (kt + 1) * BK;
            #pragma unroll
            for (int i = 0; i < 4; ++i) ax[i] = *(const float4*)(xk + i * 32 * K);
            #pragma unroll
            for (int i = 0; i < 4; ++i) wx[i] = *(const int4*)(wk + i * 32 * K);
        }
        // keep GL2 warm two tiles ahead (global_prefetch_b8, GL2-level)
        if (kt + 2 < kTiles) {
            __builtin_prefetch(xp + (kt + 2) * BK, 0, 0);
            __builtin_prefetch(wp + (kt + 2) * BK, 0, 0);
        }

        // ---- fragment loads: lane group 0 -> K 0..15, group 1 -> K 16..31.
        // Same per-lane K ordering for A and B, so any hw slot<->K permutation
        // is applied consistently to both operands and cancels in the dot. ----
        Frag a[4], b[2];
        #pragma unroll
        for (int i = 0; i < 4; ++i) {
            const unsigned int* p = &lA[(wm + i * 16 + l16) * LDS_STRIDE + half * 8];
            a[i].q[0] = *(const uint4*)(p);
            a[i].q[1] = *(const uint4*)(p + 4);
        }
        #pragma unroll
        for (int j = 0; j < 2; ++j) {
            const unsigned int* p = &lB[(wn + j * 16 + l16) * LDS_STRIDE + half * 8];
            b[j].q[0] = *(const uint4*)(p);
            b[j].q[1] = *(const uint4*)(p + 4);
        }

        // ---- 8 matrix ops per wave per K-step ----
        #pragma unroll
        for (int i = 0; i < 4; ++i)
            #pragma unroll
            for (int j = 0; j < 2; ++j)
                acc[i][j] = __builtin_amdgcn_wmma_f32_16x16x32_bf16(
                    /*neg_a=*/false, a[i].v, /*neg_b=*/false, b[j].v,
                    /*c_mod=*/(short)0, acc[i][j],
                    /*reuse_a=*/false, /*reuse_b=*/false);

        __syncthreads();
    }

    // ---- epilogue: Y = scale*acc + bias[n] ----
    // C layout: VGPR r, lane l -> M = r + 8*(l>=16), N = l%16
    #pragma unroll
    for (int i = 0; i < 4; ++i) {
        #pragma unroll
        for (int j = 0; j < 2; ++j) {
            const int gn = n0 + wn + j * 16 + l16;
            const float bn = bias[gn];
            #pragma unroll
            for (int r = 0; r < 8; ++r) {
                const int gm = m0 + wm + i * 16 + half * 8 + r;
                Y[(size_t)gm * N + gn] = __builtin_fmaf(scale, acc[i][j][r], bn);
            }
        }
    }
}

extern "C" void kernel_launch(void* const* d_in, const int* in_sizes, int n_in,
                              void* d_out, int out_size, void* d_ws, size_t ws_size,
                              hipStream_t stream) {
    const float* X    = (const float*)d_in[0];   // [4,2048,4096] fp32
    const int*   Wq   = (const int*)  d_in[1];   // [11008,4096] int32 (int8 values)
    const float* sc   = (const float*)d_in[2];   // [1]
    const float* zp   = (const float*)d_in[3];   // [1]
    const float* bias = (const float*)d_in[4];   // [11008]
    float*       Y    = (float*)d_out;           // [4,2048,11008] fp32

    const int M = 4 * 2048;                      // 8192
    const int OUT = 11008;

    dim3 grid(OUT / BN, M / BM);                 // (86, 64)
    qlinear_wmma_bf16<<<grid, 256, 0, stream>>>(X, Wq, sc, zp, bias, Y);
}
